// MultiHeadAttention_36215164240783
// MI455X (gfx1250) — matrix operand order repaired and checked by reading, not compiled, not run on hardware
//
#include <hip/hip_runtime.h>
#include <stdint.h>

#define D_MODEL 1024
#define NUM_HEADS 16
#define DK 64
#define BATCH 2
#define SEQ 2048
#define MTOK (BATCH * SEQ)   // 4096 tokens

typedef __attribute__((ext_vector_type(16))) __bf16 bf16x16;
typedef __attribute__((ext_vector_type(8)))  float  f32x8;
typedef __attribute__((ext_vector_type(4)))  float  f32x4;
typedef __attribute__((ext_vector_type(4)))  unsigned int u32x4;

struct FragU { u32x4 a, b; };   // 32 bytes == 16 bf16

__device__ __forceinline__ f32x8 zero8() {
  f32x8 z;
#pragma unroll
  for (int i = 0; i < 8; ++i) z[i] = 0.0f;
  return z;
}

// float -> bf16, round-to-nearest-even
__device__ __forceinline__ unsigned short f2bf(float f) {
  unsigned int u = __float_as_uint(f);
  u += 0x7FFFu + ((u >> 16) & 1u);
  return (unsigned short)(u >> 16);
}

__device__ __forceinline__ unsigned long long pack_bf4(f32x4 v) {
  return (unsigned long long)f2bf(v.x)
       | ((unsigned long long)f2bf(v.y) << 16)
       | ((unsigned long long)f2bf(v.z) << 32)
       | ((unsigned long long)f2bf(v.w) << 48);
}

// ---- CDNA5 async memory->LDS copy (ASYNCcnt-tracked), ISA ch.10 / §15.18 ----
// Low 32 bits of a generic pointer into __shared__ == LDS byte offset
// (flat LDS aperture truncates addr[31:0], ISA 10.2).
__device__ __forceinline__ void async_ld_b128(void* lds, const void* gaddr) {
  unsigned int loff = (unsigned int)(uintptr_t)lds;
  asm volatile("global_load_async_to_lds_b128 %0, %1, off"
               :: "v"(loff), "v"(gaddr) : "memory");
}
__device__ __forceinline__ void wait_async0() {
  asm volatile("s_wait_asynccnt 0" ::: "memory");
}

// A fragment (16x32 bf16, M x K): lane L(&15) = row M; K = hi*8 + {0..7} and hi*8 + {16..23}
__device__ __forceinline__ bf16x16 load_fragA(const unsigned short* sm, int rowBase,
                                              int kBase, int stride) {
  const int lane = (int)(threadIdx.x & 31u);
  const unsigned short* p = sm + (rowBase + (lane & 15)) * stride + kBase + (lane >> 4) * 8;
  FragU f;
  f.a = *(const u32x4*)(p);
  f.b = *(const u32x4*)(p + 16);
  return __builtin_bit_cast(bf16x16, f);
}

// B fragment (32x16 bf16, K x N): lane L(&15) = column N; K = hi*8 + {0..7} and hi*8 + {16..23}
__device__ __forceinline__ bf16x16 load_fragB(const unsigned short* sm, int colBase,
                                              int kBase, int stride) {
  const int lane = (int)(threadIdx.x & 31u);
  const unsigned short* p = sm + (colBase + (lane & 15)) * stride + kBase + (lane >> 4) * 8;
  FragU f;
  f.a = *(const u32x4*)(p);
  f.b = *(const u32x4*)(p + 16);
  return __builtin_bit_cast(bf16x16, f);
}

#define WMMA_BF16(A, B, C) \
  __builtin_amdgcn_wmma_f32_16x16x32_bf16(false, (A), false, (B), (short)0, (C), false, false)

// ---------------------------------------------------------------------------
// Kernel 1: fused convert + projection GEMM: dst = bf16(X @ W^T + bias)
//   Software-pipelined: global f32 loads for tile k+1 overlap tile k's WMMAs.
//   mode 0/1: dst layout [b,h,s,dk]; mode 2 (V): dst layout [b,h,dk,s]
// ---------------------------------------------------------------------------
#define PT_M 128
#define PT_N 128
#define PSTR 40   // padded LDS K-stride (elements); 80B rows keep 16B alignment

__global__ __launch_bounds__(256) void qkv_proj_kernel(
    const float* __restrict__ X, const float* __restrict__ W,
    const float* __restrict__ bias, unsigned short* __restrict__ dst, int mode)
{
  __shared__ unsigned short As[PT_M * PSTR];
  __shared__ unsigned short Bs[PT_N * PSTR];

  const int tid  = (int)threadIdx.x;
  const int wid  = tid >> 5;
  const int lane = tid & 31;
  const int lo16 = lane & 15;
  const int hi8  = (lane >> 4) * 8;
  const int wm   = (wid & 3) * 32;
  const int wn   = (wid >> 2) * 64;
  const int m0   = (int)blockIdx.x * PT_M;
  const int n0   = (int)blockIdx.y * PT_N;

  f32x8 acc[2][4];
#pragma unroll
  for (int i = 0; i < 2; ++i)
#pragma unroll
    for (int j = 0; j < 4; ++j) acc[i][j] = zero8();

  const int ldRow = tid >> 1;
  const int ldCol = (tid & 1) * 16;
  const float* Abase = X + (size_t)(m0 + ldRow) * D_MODEL + ldCol;
  const float* Bbase = W + (size_t)(n0 + ldRow) * D_MODEL + ldCol;
  unsigned long long* AsW = (unsigned long long*)&As[ldRow * PSTR + ldCol];
  unsigned long long* BsW = (unsigned long long*)&Bs[ldRow * PSTR + ldCol];

  // register prefetch of tile kk=0
  f32x4 ra[4], rb[4];
#pragma unroll
  for (int i = 0; i < 4; ++i) {
    ra[i] = ((const f32x4*)Abase)[i];
    rb[i] = ((const f32x4*)Bbase)[i];
  }

  for (int kk = 0; kk < D_MODEL; kk += 32) {
    __syncthreads();                       // LDS free for rewrite
#pragma unroll
    for (int i = 0; i < 4; ++i) {
      AsW[i] = pack_bf4(ra[i]);
      BsW[i] = pack_bf4(rb[i]);
    }
    __syncthreads();

    if (kk + 32 < D_MODEL) {               // prefetch next tile during compute
      const f32x4* ag = (const f32x4*)(Abase + kk + 32);
      const f32x4* bg = (const f32x4*)(Bbase + kk + 32);
#pragma unroll
      for (int i = 0; i < 4; ++i) { ra[i] = ag[i]; rb[i] = bg[i]; }
    }

    bf16x16 afr[2], bfr[4];
#pragma unroll
    for (int mt = 0; mt < 2; ++mt) afr[mt] = load_fragA(As, wm + mt * 16, 0, PSTR);
#pragma unroll
    for (int nt = 0; nt < 4; ++nt) bfr[nt] = load_fragB(Bs, wn + nt * 16, 0, PSTR);
#pragma unroll
    for (int mt = 0; mt < 2; ++mt)
#pragma unroll
      for (int nt = 0; nt < 4; ++nt)
        acc[mt][nt] = WMMA_BF16(afr[mt], bfr[nt], acc[mt][nt]);
  }

#pragma unroll
  for (int mt = 0; mt < 2; ++mt) {
#pragma unroll
    for (int nt = 0; nt < 4; ++nt) {
      const int n  = n0 + wn + nt * 16 + lo16;
      const float bn = bias[n];
      const int hh = n >> 6, d = n & 63;
#pragma unroll
      for (int r = 0; r < 8; ++r) {
        const int m = m0 + wm + mt * 16 + r + hi8;
        const int bb = m >> 11, s = m & (SEQ - 1);
        const float v = acc[mt][nt][r] + bn;
        size_t off;
        if (mode == 2) off = ((size_t)(bb * NUM_HEADS + hh) * DK + d) * SEQ + s;
        else           off = ((size_t)(bb * NUM_HEADS + hh) * SEQ + s) * DK + d;
        dst[off] = f2bf(v);
      }
    }
  }
}

// ---------------------------------------------------------------------------
// Kernel 2: flash attention per (b, h, 64-query tile). 4 waves, 16 rows each.
//   Double-buffered K/V tiles loaded with GLOBAL_LOAD_ASYNC_TO_LDS_B128:
//   next tile's async copy overlaps current tile's 16 WMMAs + softmax.
// ---------------------------------------------------------------------------
#define SSTR 72   // padded LDS stride; 144B rows keep 16B alignment

__device__ __forceinline__ void issue_kv_tile_async(
    int tid, unsigned short* KsT, unsigned short* VsT,
    const unsigned short* kg, const unsigned short* vg)
{
#pragma unroll
  for (int i = 0; i < 4; ++i) {
    const int lin = tid + i * 128;
    const int row = lin >> 3, cg = (lin & 7) * 8;
    async_ld_b128(&KsT[row * SSTR + cg], kg + row * DK + cg);
    async_ld_b128(&VsT[row * SSTR + cg], vg + (size_t)row * SEQ + cg);
  }
}

__global__ __launch_bounds__(128) void attention_kernel(
    const unsigned short* __restrict__ qws,
    const unsigned short* __restrict__ kws,
    const unsigned short* __restrict__ vws,
    unsigned short* __restrict__ ows)
{
  __shared__ unsigned short Qs[64 * SSTR];
  __shared__ unsigned short Ks[2][64 * SSTR];
  __shared__ unsigned short Vs[2][64 * SSTR];
  __shared__ unsigned short Ps[64 * SSTR];

  const int tid  = (int)threadIdx.x;
  const int wid  = tid >> 5;
  const int lane = tid & 31;
  const int lo16 = lane & 15;
  const int hi8  = (lane >> 4) * 8;

  const int qbase = (int)blockIdx.x * 64;
  const int h     = (int)blockIdx.y;
  const int bb    = (int)blockIdx.z;
  const size_t bh = (size_t)bb * NUM_HEADS + h;

  const unsigned short* qg = qws + (bh * SEQ + qbase) * DK;
  const unsigned short* kbase = kws + bh * SEQ * DK;
  const unsigned short* vbase = vws + bh * DK * SEQ;

  // async-copy Q tile + first K/V tile into LDS
#pragma unroll
  for (int i = 0; i < 4; ++i) {
    const int lin = tid + i * 128;
    const int row = lin >> 3, cg = (lin & 7) * 8;
    async_ld_b128(&Qs[row * SSTR + cg], qg + row * DK + cg);
  }
  issue_kv_tile_async(tid, Ks[0], Vs[0], kbase, vbase);
  wait_async0();
  __syncthreads();

  f32x8 Oacc[4];
#pragma unroll
  for (int dt = 0; dt < 4; ++dt) Oacc[dt] = zero8();
  float mrun[8], lrun[8];
#pragma unroll
  for (int r = 0; r < 8; ++r) { mrun[r] = -3.0e38f; lrun[r] = 0.0f; }
  const float scl = 0.125f;   // 1/sqrt(dk)

  for (int kt = 0; kt < SEQ; kt += 64) {
    const int cur = (kt >> 6) & 1;
    if (kt + 64 < SEQ)   // prefetch next tile; overlaps all compute below
      issue_kv_tile_async(tid, Ks[cur ^ 1], Vs[cur ^ 1],
                          kbase + (size_t)(kt + 64) * DK, vbase + (kt + 64));
    const unsigned short* KsC = Ks[cur];
    const unsigned short* VsC = Vs[cur];

    // S = Q_tile (16x64) @ K_tile^T (64x64): B columns = K rows
    f32x8 sacc[4];
#pragma unroll
    for (int nt = 0; nt < 4; ++nt) sacc[nt] = zero8();
#pragma unroll
    for (int ks = 0; ks < DK; ks += 32) {
      bf16x16 aq = load_fragA(Qs, wid * 16, ks, SSTR);
#pragma unroll
      for (int nt = 0; nt < 4; ++nt) {
        bf16x16 bk = load_fragB(KsC, nt * 16, ks, SSTR);
        sacc[nt] = WMMA_BF16(aq, bk, sacc[nt]);
      }
    }

    // online softmax; row (r, hi) stats replicated over 16 lanes of each half
#pragma unroll
    for (int r = 0; r < 8; ++r) {
      float s0 = sacc[0][r] * scl, s1 = sacc[1][r] * scl;
      float s2 = sacc[2][r] * scl, s3 = sacc[3][r] * scl;
      float mx = fmaxf(fmaxf(s0, s1), fmaxf(s2, s3));
      mx = fmaxf(mx, __shfl_xor(mx, 1, 32));
      mx = fmaxf(mx, __shfl_xor(mx, 2, 32));
      mx = fmaxf(mx, __shfl_xor(mx, 4, 32));
      mx = fmaxf(mx, __shfl_xor(mx, 8, 32));
      const float mnew = fmaxf(mrun[r], mx);
      const float corr = __expf(mrun[r] - mnew);
      const float p0 = __expf(s0 - mnew);
      const float p1 = __expf(s1 - mnew);
      const float p2 = __expf(s2 - mnew);
      const float p3 = __expf(s3 - mnew);
      float rs = p0 + p1 + p2 + p3;
      rs += __shfl_xor(rs, 1, 32);
      rs += __shfl_xor(rs, 2, 32);
      rs += __shfl_xor(rs, 4, 32);
      rs += __shfl_xor(rs, 8, 32);
      lrun[r] = lrun[r] * corr + rs;
      mrun[r] = mnew;
#pragma unroll
      for (int dt = 0; dt < 4; ++dt) Oacc[dt][r] *= corr;
      const int pm = wid * 16 + r + hi8;
      Ps[pm * SSTR +  0 + lo16] = f2bf(p0);
      Ps[pm * SSTR + 16 + lo16] = f2bf(p1);
      Ps[pm * SSTR + 32 + lo16] = f2bf(p2);
      Ps[pm * SSTR + 48 + lo16] = f2bf(p3);
    }
    __syncthreads();   // Ps visible (async prefetch still in flight)

    // O += P_tile (16x64) @ V_tile (64x64): B columns = Vs rows (v transposed)
#pragma unroll
    for (int ks = 0; ks < 64; ks += 32) {
      bf16x16 ap = load_fragA(Ps, wid * 16, ks, SSTR);
#pragma unroll
      for (int dt = 0; dt < 4; ++dt) {
        bf16x16 bvf = load_fragB(VsC, dt * 16, ks, SSTR);
        Oacc[dt] = WMMA_BF16(ap, bvf, Oacc[dt]);
      }
    }

    wait_async0();     // next tile landed
    __syncthreads();
  }

#pragma unroll
  for (int r = 0; r < 8; ++r) {
    const float inv = 1.0f / lrun[r];
    const int s = qbase + wid * 16 + r + hi8;
    const size_t rowoff = ((size_t)bb * SEQ + s) * D_MODEL + h * DK;
#pragma unroll
    for (int dt = 0; dt < 4; ++dt)
      ows[rowoff + dt * 16 + lo16] = f2bf(Oacc[dt][r] * inv);
  }
}

// ---------------------------------------------------------------------------
// Kernel 3: output projection: out_f32 = attn_bf16 @ Wo^T + bo
//   A tile (already bf16) copied with async-to-LDS; B pack + next-B global
//   loads overlap the async copy / WMMAs.
// ---------------------------------------------------------------------------
__global__ __launch_bounds__(256) void out_proj_kernel(
    const unsigned short* __restrict__ Abf, const float* __restrict__ W,
    const float* __restrict__ bias, float* __restrict__ out)
{
  __shared__ unsigned short As[PT_M * PSTR];
  __shared__ unsigned short Bs[PT_N * PSTR];

  const int tid  = (int)threadIdx.x;
  const int wid  = tid >> 5;
  const int lane = tid & 31;
  const int lo16 = lane & 15;
  const int hi8  = (lane >> 4) * 8;
  const int wm   = (wid & 3) * 32;
  const int wn   = (wid >> 2) * 64;
  const int m0   = (int)blockIdx.x * PT_M;
  const int n0   = (int)blockIdx.y * PT_N;

  f32x8 acc[2][4];
#pragma unroll
  for (int i = 0; i < 2; ++i)
#pragma unroll
    for (int j = 0; j < 4; ++j) acc[i][j] = zero8();

  const int ldRow = tid >> 1;
  const int ldCol = (tid & 1) * 16;
  const unsigned short* Abase = Abf + (size_t)(m0 + ldRow) * D_MODEL + ldCol;
  const float* Bbase = W + (size_t)(n0 + ldRow) * D_MODEL + ldCol;
  unsigned short* AsP = &As[ldRow * PSTR + ldCol];
  unsigned long long* BsW = (unsigned long long*)&Bs[ldRow * PSTR + ldCol];

  f32x4 rb[4];
#pragma unroll
  for (int i = 0; i < 4; ++i) rb[i] = ((const f32x4*)Bbase)[i];

  for (int kk = 0; kk < D_MODEL; kk += 32) {
    __syncthreads();                       // LDS free
    async_ld_b128(AsP,     Abase + kk);        // A tile: memory -> LDS direct
    async_ld_b128(AsP + 8, Abase + kk + 8);
#pragma unroll
    for (int i = 0; i < 4; ++i) BsW[i] = pack_bf4(rb[i]);
    if (kk + 32 < D_MODEL) {
      const f32x4* bg = (const f32x4*)(Bbase + kk + 32);
#pragma unroll
      for (int i = 0; i < 4; ++i) rb[i] = bg[i];
    }
    wait_async0();
    __syncthreads();

    bf16x16 afr[2], bfr[4];
#pragma unroll
    for (int mt = 0; mt < 2; ++mt) afr[mt] = load_fragA(As, wm + mt * 16, 0, PSTR);
#pragma unroll
    for (int nt = 0; nt < 4; ++nt) bfr[nt] = load_fragB(Bs, wn + nt * 16, 0, PSTR);
#pragma unroll
    for (int mt = 0; mt < 2; ++mt)
#pragma unroll
      for (int nt = 0; nt < 4; ++nt)
        acc[mt][nt] = WMMA_BF16(afr[mt], bfr[nt], acc[mt][nt]);
  }

#pragma unroll
  for (int mt = 0; mt < 2; ++mt) {
#pragma unroll
    for (int nt = 0; nt < 4; ++nt) {
      const int n = n0 + wn + nt * 16 + lo16;
      const float bn = bias[n];
#pragma unroll
      for (int r = 0; r < 8; ++r) {
        const int m = m0 + wm + mt * 16 + r + hi8;
        out[(size_t)m * D_MODEL + n] = acc[mt][nt][r] + bn;
      }
    }
  }
}

// ---------------------------------------------------------------------------
extern "C" void kernel_launch(void* const* d_in, const int* in_sizes, int n_in,
                              void* d_out, int out_size, void* d_ws, size_t ws_size,
                              hipStream_t stream) {
  (void)in_sizes; (void)n_in; (void)out_size; (void)ws_size;
  const float* Q  = (const float*)d_in[0];
  const float* K  = (const float*)d_in[1];
  const float* V  = (const float*)d_in[2];
  const float* Wq = (const float*)d_in[3];
  const float* bq = (const float*)d_in[4];
  const float* Wk = (const float*)d_in[5];
  const float* bk = (const float*)d_in[6];
  const float* Wv = (const float*)d_in[7];
  const float* bv = (const float*)d_in[8];
  const float* Wo = (const float*)d_in[9];
  const float* bo = (const float*)d_in[10];

  unsigned short* qws = (unsigned short*)d_ws;                 // 8 MB each
  unsigned short* kws = qws + (size_t)MTOK * D_MODEL;
  unsigned short* vws = kws + (size_t)MTOK * D_MODEL;
  unsigned short* aws = vws + (size_t)MTOK * D_MODEL;          // total 32 MB

  dim3 gb(MTOK / PT_M, D_MODEL / PT_N);   // 32 x 8
  qkv_proj_kernel<<<gb, 256, 0, stream>>>(Q, Wq, bq, qws, 0);
  qkv_proj_kernel<<<gb, 256, 0, stream>>>(K, Wk, bk, kws, 0);
  qkv_proj_kernel<<<gb, 256, 0, stream>>>(V, Wv, bv, vws, 2);
  attention_kernel<<<dim3(SEQ / 64, NUM_HEADS, BATCH), 128, 0, stream>>>(qws, kws, vws, aws);
  out_proj_kernel<<<gb, 256, 0, stream>>>(aws, Wo, bo, (float*)d_out);
}
